// SelfAttention_74174085202165
// MI455X (gfx1250) — matrix operand order repaired and checked by reading, not compiled, not run on hardware
//
#include <hip/hip_runtime.h>
#include <hip/hip_bf16.h>
#include <math.h>

// ---------------------------------------------------------------------------
// CDNA5 (gfx1250, wave32) WMMA self-attention block.
//   per-head linear (64x64) -> flash softmax attention -> Wl -> LN -> +q
// GEMMs: v_wmma_f32_16x16x32_bf16. K/V and GEMM A-tiles staged into LDS with
// the Tensor Data Mover (tensor_load_to_lds), double-buffered, synchronized
// with s_wait_tensorcnt + workgroup barriers.
// ---------------------------------------------------------------------------

#define HEADS    16
#define HEAD_DIM 64
#define DMODEL   1024
#define SEQ      2048
#define BATCH    4
#define ROWS     (SEQ * BATCH)          // 8192 token rows of DMODEL
#define HROWS    (ROWS * HEADS)         // 131072 head rows of 64

typedef __attribute__((ext_vector_type(16))) __bf16 v16bf;
typedef __attribute__((ext_vector_type(8)))  float  v8f;
typedef unsigned int u32x4 __attribute__((ext_vector_type(4)));
typedef int          i32x4 __attribute__((ext_vector_type(4)));
typedef int          i32x8 __attribute__((ext_vector_type(8)));

__device__ __forceinline__ v8f wmma_bf16(v16bf a, v16bf b, v8f c) {
  return __builtin_amdgcn_wmma_f32_16x16x32_bf16(false, a, false, b,
                                                 (short)0, c, false, false);
}

// Low 32 bits of a generic pointer to __shared__ = LDS byte offset
__device__ __forceinline__ unsigned lds_off(const void* p) {
  return (unsigned)(unsigned long long)p;
}

// ---------------------------------------------------------------------------
// TDM: 2D tile (rows x cols elements, 2-byte data) global -> LDS.
// D# per ISA 08_async_tensor Sec. 8.3/8.4. Issued by one wave; tracked with
// TENSORcnt. tensor_dims set huge so no OOB clipping occurs.
// ---------------------------------------------------------------------------
__device__ __forceinline__ void tdm_load_2d(const __bf16* gsrc, unsigned ldsa,
                                            int rows, int cols,
                                            int row_stride_elems) {
  const unsigned long long ga = (unsigned long long)gsrc;
  const unsigned t0 = 0x7FFFFFFFu, t1 = 0x7FFFFFFFu;

  u32x4 g0 = {};
  g0[0] = 1u;                                              // count=1 (user D#)
  g0[1] = ldsa;                                            // lds_addr
  g0[2] = (unsigned)(ga & 0xFFFFFFFFu);                    // global_addr lo
  g0[3] = (unsigned)((ga >> 32) & 0x1FFFFFFu) | (2u << 30); // addr hi | type=2

  i32x8 g1 = {};
  g1[0] = (int)(1u << 16);                                 // data_size=1 (2B)
  g1[1] = (int)((t0 & 0xFFFFu) << 16);                     // tensor_dim0 lo16
  g1[2] = (int)(((t0 >> 16) & 0xFFFFu) | ((t1 & 0xFFFFu) << 16));
  g1[3] = (int)(((t1 >> 16) & 0xFFFFu) | ((unsigned)cols << 16)); // tile_dim0
  g1[4] = (int)((unsigned)rows & 0xFFFFu);                 // tile_dim1
  g1[5] = (int)(unsigned)row_stride_elems;                 // dim0_stride lo32
  g1[6] = 0;
  g1[7] = 0;

  i32x4 z4 = {};
#if __clang_major__ >= 23
  i32x8 z8 = {};
  __builtin_amdgcn_tensor_load_to_lds(g0, g1, z4, z4, z8, 0);
#else
  __builtin_amdgcn_tensor_load_to_lds(g0, g1, z4, z4, 0);
#endif
}

// ---------------------------------------------------------------------------
// Fragment loaders (contiguous k order used by this file; A and B match).
// A 16x32 bf16: M = lane&15, K = i + 16*(lane>=16)
// B 32x16 bf16: N = lane&15, K = i + 16*(lane>=16)
// ---------------------------------------------------------------------------
template <typename SRC>
__device__ __forceinline__ v16bf load_a_frag(const SRC* base, int stride) {
  const int m    = threadIdx.x & 15;
  const int koff = ((threadIdx.x >> 4) & 1) << 4;
  v16bf a;
#pragma unroll
  for (int i = 0; i < 16; ++i) {
    const int k = koff + i;
    a[i] = (__bf16)(float)base[m * stride + k];
  }
  return a;
}

// B[k][n] = src[n*stride + k]
template <typename SRC>
__device__ __forceinline__ v16bf load_b_frag_nk(const SRC* base, int stride) {
  const int n    = threadIdx.x & 15;
  const int koff = ((threadIdx.x >> 4) & 1) << 4;
  v16bf b;
#pragma unroll
  for (int i = 0; i < 16; ++i)
    b[i] = (__bf16)(float)base[n * stride + (koff + i)];
  return b;
}

// B[k][n] = src[k*stride + n]
template <typename SRC>
__device__ __forceinline__ v16bf load_b_frag_kn(const SRC* base, int stride) {
  const int n    = threadIdx.x & 15;
  const int koff = ((threadIdx.x >> 4) & 1) << 4;
  v16bf b;
#pragma unroll
  for (int i = 0; i < 16; ++i)
    b[i] = (__bf16)(float)base[(long)(koff + i) * stride + n];
  return b;
}

// ---------------------------------------------------------------------------
// Kernel 0: one-shot f32 -> bf16 conversion (Wl)
// ---------------------------------------------------------------------------
__global__ void __launch_bounds__(256) cvt_kernel(const float* __restrict__ src,
                                                  __bf16* __restrict__ dst) {
  const long i = ((long)blockIdx.x * 256 + threadIdx.x) * 4;
  const float4 f = *(const float4*)(src + i);
  dst[i + 0] = (__bf16)f.x;
  dst[i + 1] = (__bf16)f.y;
  dst[i + 2] = (__bf16)f.z;
  dst[i + 3] = (__bf16)f.w;
}

// ---------------------------------------------------------------------------
// Kernel 1: per-head projection  y[m,e] = sum_d x[m,d] * W[e,d]
// x viewed as [HROWS, 64]; one wave -> 16x64 tile (8 WMMAs). Q gets 1/32.
// ---------------------------------------------------------------------------
__global__ void __launch_bounds__(128) proj_kernel(const float*  __restrict__ x,
                                                   const float*  __restrict__ W,
                                                   __bf16*       __restrict__ out,
                                                   float scale) {
  const int  wave  = threadIdx.x >> 5;
  const long wrow0 = ((long)blockIdx.x * 4 + wave) * 16;

  const v16bf A0 = load_a_frag(x + wrow0 * HEAD_DIM,      HEAD_DIM);
  const v16bf A1 = load_a_frag(x + wrow0 * HEAD_DIM + 32, HEAD_DIM);

  v8f acc[4] = {};
#pragma unroll
  for (int nt = 0; nt < 4; ++nt) {
    acc[nt] = wmma_bf16(A0, load_b_frag_nk(W + nt * 16 * 64,      64), acc[nt]);
    acc[nt] = wmma_bf16(A1, load_b_frag_nk(W + nt * 16 * 64 + 32, 64), acc[nt]);
  }

  const int n    = threadIdx.x & 15;
  const int half = (threadIdx.x >> 4) & 1;
#pragma unroll
  for (int nt = 0; nt < 4; ++nt)
#pragma unroll
    for (int r = 0; r < 8; ++r) {
      const long row = wrow0 + r + 8 * half;
      out[row * HEAD_DIM + nt * 16 + n] = (__bf16)(acc[nt][r] * scale);
    }
}

// ---------------------------------------------------------------------------
// Kernel 2: flash attention per (b,h). WG = 4 waves x 16 query rows.
// K/V chunks (32 x 64 bf16) TDM-staged into double-buffered LDS, shared by
// all 4 waves. Online softmax; P re-laid out D-frag -> A-frag through LDS.
// ---------------------------------------------------------------------------
__global__ void __launch_bounds__(128) attn_kernel(const __bf16* __restrict__ Q,
                                                   const __bf16* __restrict__ K,
                                                   const __bf16* __restrict__ V,
                                                   __bf16*       __restrict__ O) {
  __shared__ __bf16 kbuf[2][32][64];
  __shared__ __bf16 vbuf[2][32][64];
  __shared__ __bf16 plds[4][16][32];

  const int wave = threadIdx.x >> 5;
  const int lane = threadIdx.x & 31;
  const int n    = lane & 15;
  const int half = lane >> 4;

  const int  b       = blockIdx.y >> 4;
  const int  h       = blockIdx.y & 15;
  const long base_bh = (long)b * SEQ * DMODEL + (long)h * HEAD_DIM;
  const int  q0      = blockIdx.x * 64 + wave * 16;

  const __bf16* Kbase = K + base_bh;
  const __bf16* Vbase = V + base_bh;

  // prime chunk 0
  if (threadIdx.x < 32) {
    tdm_load_2d(Kbase, lds_off(&kbuf[0][0][0]), 32, 64, DMODEL);
    tdm_load_2d(Vbase, lds_off(&vbuf[0][0][0]), 32, 64, DMODEL);
  }

  const v16bf Aq0 = load_a_frag(Q + base_bh + (long)q0 * DMODEL,      DMODEL);
  const v16bf Aq1 = load_a_frag(Q + base_bh + (long)q0 * DMODEL + 32, DMODEL);

  float mrow[8], lrow[8];
  v8f   acc[4] = {};
#pragma unroll
  for (int r = 0; r < 8; ++r) { mrow[r] = -1e30f; lrow[r] = 0.0f; }

  const int NCHUNK = SEQ / 32;
  for (int c = 0; c < NCHUNK; ++c) {
    const int p = c & 1;
    if (threadIdx.x < 32) __builtin_amdgcn_s_wait_tensorcnt((short)0);
    __syncthreads();                              // buffers[p] published
    if (threadIdx.x < 32 && (c + 1) < NCHUNK) {   // overlap next chunk DMA
      tdm_load_2d(Kbase + (long)(c + 1) * 32 * DMODEL,
                  lds_off(&kbuf[1 - p][0][0]), 32, 64, DMODEL);
      tdm_load_2d(Vbase + (long)(c + 1) * 32 * DMODEL,
                  lds_off(&vbuf[1 - p][0][0]), 32, 64, DMODEL);
    }

    // ---- scores chunk: 16(q) x 32(k), two 16x16 D fragments -------------
    v8f s[2] = {};
#pragma unroll
    for (int t = 0; t < 2; ++t) {
      s[t] = wmma_bf16(Aq0, load_b_frag_nk(&kbuf[p][t * 16][0],      64), s[t]);
      s[t] = wmma_bf16(Aq1, load_b_frag_nk(&kbuf[p][t * 16][0] + 32, 64), s[t]);
    }

    // ---- online softmax update -----------------------------------------
    float cmax[8];
#pragma unroll
    for (int r = 0; r < 8; ++r) cmax[r] = fmaxf(s[0][r], s[1][r]);
#pragma unroll
    for (int mask = 1; mask <= 8; mask <<= 1)
#pragma unroll
      for (int r = 0; r < 8; ++r)
        cmax[r] = fmaxf(cmax[r], __shfl_xor(cmax[r], mask, 32));

    float alpha[8];
#pragma unroll
    for (int r = 0; r < 8; ++r) {
      const float mn = fmaxf(mrow[r], cmax[r]);
      alpha[r] = __expf(mrow[r] - mn);
      mrow[r]  = mn;
    }
#pragma unroll
    for (int t = 0; t < 2; ++t)
#pragma unroll
      for (int r = 0; r < 8; ++r)
        s[t][r] = __expf(s[t][r] - mrow[r]);

    float csum[8];
#pragma unroll
    for (int r = 0; r < 8; ++r) csum[r] = s[0][r] + s[1][r];
#pragma unroll
    for (int mask = 1; mask <= 8; mask <<= 1)
#pragma unroll
      for (int r = 0; r < 8; ++r)
        csum[r] += __shfl_xor(csum[r], mask, 32);

#pragma unroll
    for (int r = 0; r < 8; ++r) lrow[r] = lrow[r] * alpha[r] + csum[r];
#pragma unroll
    for (int nt = 0; nt < 4; ++nt)
#pragma unroll
      for (int r = 0; r < 8; ++r) acc[nt][r] *= alpha[r];

    // ---- P: D-fragment -> LDS -> A-fragment (wave-private tile) ---------
#pragma unroll
    for (int t = 0; t < 2; ++t)
#pragma unroll
      for (int r = 0; r < 8; ++r)
        plds[wave][r + 8 * half][t * 16 + n] = (__bf16)s[t][r];
    asm volatile("s_wait_dscnt 0x0" ::: "memory");
    const v16bf pA = load_a_frag(&plds[wave][0][0], 32);

    // ---- acc += P @ V_chunk (4 WMMAs, K=32) -----------------------------
#pragma unroll
    for (int nt = 0; nt < 4; ++nt)
      acc[nt] = wmma_bf16(pA, load_b_frag_kn(&vbuf[p][0][0] + nt * 16, 64),
                          acc[nt]);

    __syncthreads();                              // done reading buffers[p]
  }

  // ---- normalize, emit O[(b*SEQ+q)*1024 + h*64 + d] ----------------------
#pragma unroll
  for (int nt = 0; nt < 4; ++nt)
#pragma unroll
    for (int r = 0; r < 8; ++r) {
      const long row = q0 + r + 8 * half;
      O[base_bh + row * DMODEL + nt * 16 + n] = (__bf16)(acc[nt][r] / lrow[r]);
    }
}

// ---------------------------------------------------------------------------
// Kernel 3: lin = O @ Wl^T ; LayerNorm over DMODEL ; out = q + LN(lin)
// WG = 16 rows x 1024 cols; wave owns 256 cols. A-tile (16x32) TDM-staged
// into double-buffered LDS (shared by 4 waves); Wl pre-converted to bf16.
// ---------------------------------------------------------------------------
__global__ void __launch_bounds__(128) outln_kernel(const __bf16* __restrict__ Ob,
                                                    const __bf16* __restrict__ Wlb,
                                                    const float*  __restrict__ gamma,
                                                    const float*  __restrict__ beta,
                                                    const float*  __restrict__ qin,
                                                    float*        __restrict__ out) {
  __shared__ __bf16 abuf[2][16][32];
  __shared__ float red_s[4][16];
  __shared__ float red_q[4][16];

  const int wave = threadIdx.x >> 5;
  const int lane = threadIdx.x & 31;
  const int n    = lane & 15;
  const int half = lane >> 4;

  const long row0 = (long)blockIdx.x * 16;
  const int  n0   = wave * 256;

  if (threadIdx.x < 32)
    tdm_load_2d(Ob + row0 * DMODEL, lds_off(&abuf[0][0][0]), 16, 32, DMODEL);

  v8f acc[16] = {};
  for (int k0 = 0; k0 < DMODEL; k0 += 32) {
    const int p = (k0 >> 5) & 1;
    if (threadIdx.x < 32) __builtin_amdgcn_s_wait_tensorcnt((short)0);
    __syncthreads();
    if (threadIdx.x < 32 && (k0 + 32) < DMODEL)
      tdm_load_2d(Ob + row0 * DMODEL + k0 + 32, lds_off(&abuf[1 - p][0][0]),
                  16, 32, DMODEL);

    __builtin_prefetch(Wlb + (long)n0 * DMODEL + k0 + 32, 0, 1);

    const v16bf A = load_a_frag(&abuf[p][0][0], 32);
#pragma unroll
    for (int nt = 0; nt < 16; ++nt) {
      const v16bf B =
          load_b_frag_nk(Wlb + (long)(n0 + nt * 16) * DMODEL + k0, DMODEL);
      acc[nt] = wmma_bf16(A, B, acc[nt]);
    }
    __syncthreads();
  }

  // ---- per-row sums over this wave's 256 columns ------------------------
  float ls[8] = {}, lq[8] = {};
#pragma unroll
  for (int nt = 0; nt < 16; ++nt)
#pragma unroll
    for (int r = 0; r < 8; ++r) {
      const float v = acc[nt][r];
      ls[r] += v;
      lq[r] += v * v;
    }
#pragma unroll
  for (int mask = 1; mask <= 8; mask <<= 1)
#pragma unroll
    for (int r = 0; r < 8; ++r) {
      ls[r] += __shfl_xor(ls[r], mask, 32);
      lq[r] += __shfl_xor(lq[r], mask, 32);
    }
  if (n == 0) {
#pragma unroll
    for (int r = 0; r < 8; ++r) {
      red_s[wave][half * 8 + r] = ls[r];
      red_q[wave][half * 8 + r] = lq[r];
    }
  }
  __syncthreads();

  float mean[8], rstd[8];
#pragma unroll
  for (int r = 0; r < 8; ++r) {
    const int rl = half * 8 + r;
    float ts = 0.0f, tq = 0.0f;
#pragma unroll
    for (int w = 0; w < 4; ++w) { ts += red_s[w][rl]; tq += red_q[w][rl]; }
    const float mu  = ts * (1.0f / DMODEL);
    const float var = tq * (1.0f / DMODEL) - mu * mu;
    mean[r] = mu;
    rstd[r] = rsqrtf(var + 1e-5f);
  }

#pragma unroll
  for (int nt = 0; nt < 16; ++nt)
#pragma unroll
    for (int r = 0; r < 8; ++r) {
      const long row = row0 + r + 8 * half;
      const int  col = n0 + nt * 16 + n;
      const long idx = row * DMODEL + col;
      out[idx] = (acc[nt][r] - mean[r]) * rstd[r] * gamma[col] + beta[col] + qin[idx];
    }
}

// ---------------------------------------------------------------------------
extern "C" void kernel_launch(void* const* d_in, const int* in_sizes, int n_in,
                              void* d_out, int out_size, void* d_ws, size_t ws_size,
                              hipStream_t stream) {
  (void)in_sizes; (void)n_in; (void)out_size; (void)ws_size;
  const float* q     = (const float*)d_in[0];
  const float* k     = (const float*)d_in[1];
  const float* v     = (const float*)d_in[2];
  const float* Wq    = (const float*)d_in[3];
  const float* Wk    = (const float*)d_in[4];
  const float* Wv    = (const float*)d_in[5];
  const float* Wl    = (const float*)d_in[6];
  const float* gamma = (const float*)d_in[7];
  const float* beta  = (const float*)d_in[8];

  const size_t SLAB = (size_t)ROWS * DMODEL * sizeof(__bf16);   // 16 MiB
  char* ws = (char*)d_ws;
  __bf16* Qb  = (__bf16*)(ws + 0 * SLAB);
  __bf16* Kb  = (__bf16*)(ws + 1 * SLAB);
  __bf16* Vb  = (__bf16*)(ws + 2 * SLAB);
  __bf16* Ob  = (__bf16*)(ws + 3 * SLAB);
  __bf16* Wlb = (__bf16*)(ws + 4 * SLAB);                       // 2 MiB

  cvt_kernel<<<(DMODEL * DMODEL) / 1024, 256, 0, stream>>>(Wl, Wlb);

  // scores divided by sqrt(DMODEL)=32: fold into Q (exact in bf16)
  proj_kernel<<<HROWS / 64, 128, 0, stream>>>(q, Wq, Qb, 1.0f / 32.0f);
  proj_kernel<<<HROWS / 64, 128, 0, stream>>>(k, Wk, Kb, 1.0f);
  proj_kernel<<<HROWS / 64, 128, 0, stream>>>(v, Wv, Vb, 1.0f);

  attn_kernel<<<dim3(SEQ / 64, BATCH * HEADS), 128, 0, stream>>>(Qb, Kb, Vb, Ob);

  outln_kernel<<<ROWS / 16, 128, 0, stream>>>(Ob, Wlb, gamma, beta, q,
                                              (float*)d_out);
}
